// VariationalSeq2Seq_baseline_90726889160847
// MI455X (gfx1250) — matrix operand order repaired and checked by reading, not compiled, not run on hardware
//
#include <hip/hip_runtime.h>
#include <cstddef>

typedef __attribute__((ext_vector_type(16))) _Float16 v16h;
typedef __attribute__((ext_vector_type(8)))  float    v8f;

#define SEQL 512
#define NBAT 1024
#define OBSD 64

// ---- f16 LDS pool layout (offsets in halfs) ----
#define OF_FRAG 0                   // 46 B-fragments, 512 halfs each
#define FRAG(i) (OF_FRAG + (i) * 512)
#define OF_E    (46 * 512)          // e3 / dec-emb1 staging fragment
#define OF_E1   (OF_E + 1024)       // 2 frags (e1, K=64)
#define OF_X    (OF_E1 + 1024)
#define OF_X2   (OF_X + 512)
#define OF_OUT  (OF_X2 + 512)
#define OF_H    (OF_OUT + 512)      // 4 GRU states: l0f,l0b,l1f,l1b
#define SMEM_HALFS (OF_H + 4 * 512)

// ---- bias offsets in f32 LDS array ----
#define BO_BE1 0
#define BO_BE2 64
#define BO_BE3 74
#define BO_EG0 84
#define BO_EG1 204
#define BO_BD1 324
#define BO_BD2 334
#define BO_DG0 344
#define BO_DG1 464
#define BO_BOUT 584

struct KParams {
    const float* p[45];
    float* out;
    void* e3ws;   // [64][512][512] f16 fragment tiles
    void* ysws;   // [64][512][512] f16 fragment tiles
};

__device__ __forceinline__ v8f wmma16(v16h a, v16h b, v8f c) {
    return __builtin_amdgcn_wmma_f32_16x16x32_f16(false, a, false, b, (short)0, c, false, false);
}

// fast sigmoid/tanh: v_exp_f32 + v_rcp_f32 (saturate correctly at +-inf)
__device__ __forceinline__ float sigmoid_f(float x) {
    return __builtin_amdgcn_rcpf(1.f + __expf(-x));
}
__device__ __forceinline__ float tanh_f(float x) {
    return 1.f - 2.f * __builtin_amdgcn_rcpf(1.f + __expf(2.f * x));
}

// position of element (m,k) inside a 16x32 f16 A-fragment stored as [lane][h]
__device__ __forceinline__ int fpos(int m, int k) {
    const int grp = (k >> 3) & 1;
    const int h = k - 8 * grp - 8 * ((k >> 4) & 1);
    return (m + 16 * grp) * 16 + h;
}

// contiguous per-lane fragment load (A or B operand): 32 bytes per lane
__device__ __forceinline__ v16h load_frag(const _Float16* sm, int off) {
    const int lane = threadIdx.x;
    v16h r;
#pragma unroll
    for (int h = 0; h < 16; ++h) r[h] = sm[off + lane * 16 + h];
    return r;
}

// gate D tile (+preloaded bias) -> row-major f32 LDS [16][32], N=30
__device__ __forceinline__ void store_gate(float* buf, v8f d, int nt, float bv) {
    const int lane = threadIdx.x;
    const int n = nt * 16 + (lane & 15);
    const int mb = (lane >= 16) ? 8 : 0;
    if (n < 30) {
#pragma unroll
        for (int r = 0; r < 8; ++r) buf[(mb + r) * 32 + n] = d[r] + bv;
    }
}

// D tile (+preloaded bias, activation) stored into A-fragment layout
__device__ __forceinline__ void store_frag_act(_Float16* sm, int off, v8f d, int nt,
                                               float bv, int N, int act) {
    const int lane = threadIdx.x;
    const int n = nt * 16 + (lane & 15);
    const int mb = (lane >= 16) ? 8 : 0;
    if (n < N) {
        const int base = off + (n >> 5) * 512 + fpos(mb, n & 31);
#pragma unroll
        for (int r = 0; r < 8; ++r) {
            float x = d[r] + bv;
            if (act == 1) x = (x > 0.f) ? x : 0.01f * x;   // LeakyReLU(0.01)
            else if (act == 2) x = tanh_f(x);
            sm[base + r * 16] = (_Float16)x;
        }
    }
}

// Build B-fragment of W^T (W is [N,K] torch layout)
__device__ __forceinline__ void stage_w(_Float16* sm, int off, const float* W,
                                        int N, int K, int nt, int kt) {
    const int lane = threadIdx.x;
    const int n = nt * 16 + (lane & 15);
    const int kbase = kt * 32 + ((lane >= 16) ? 8 : 0);
#pragma unroll
    for (int h = 0; h < 16; ++h) {
        const int k = kbase + ((h < 8) ? h : h + 8);
        const float w = (n < N && k < K) ? W[n * K + k] : 0.0f;
        sm[off + lane * 16 + h] = (_Float16)w;
    }
}

__device__ __forceinline__ void copyf(float* dst, const float* src, int n) {
    for (int i = threadIdx.x; i < n; i += 32) dst[i] = src[i];
}

__device__ __forceinline__ float bias_hi30(const float* b) {   // bias[16+nc] guarded vs 30
    const int n = 16 + (threadIdx.x & 15);
    return (n < 30) ? b[n] : 0.f;
}

#define DST_NONE 0
#define DST_LDS 1
#define DST_GLB 2

// GRU cell for a 16-row tile. x operand passed in registers; h state in LDS.
template <int DSTK>
__device__ __forceinline__ void gru_cell(_Float16* sm, v16h ax, int hoff,
                                         v16h wi0, v16h wi1, v16h wh0, v16h wh1,
                                         float bi0, float bi1, float bh0, float bh1,
                                         float* sgi, float* sgh,
                                         int ldstOff, int dstCol, _Float16* gdst) {
    v8f z = {};
    v16h ah = load_frag(sm, hoff);
    v8f gi0 = wmma16(ax, wi0, z);
    v8f gi1 = wmma16(ax, wi1, z);
    v8f gh0 = wmma16(ah, wh0, z);
    v8f gh1 = wmma16(ah, wh1, z);
    store_gate(sgi, gi0, 0, bi0);
    store_gate(sgi, gi1, 1, bi1);
    store_gate(sgh, gh0, 0, bh0);
    store_gate(sgh, gh1, 1, bh1);
    __syncthreads();
#pragma unroll
    for (int it = 0; it < 5; ++it) {               // 160 = 16 rows x 10 features
        const int idx = (int)threadIdx.x + it * 32;
        const int b = idx / 10, j = idx - b * 10;
        const float ir = sgi[b * 32 + j], iz = sgi[b * 32 + j + 10], in_ = sgi[b * 32 + j + 20];
        const float hr = sgh[b * 32 + j], hz = sgh[b * 32 + j + 10], hn = sgh[b * 32 + j + 20];
        const float r  = sigmoid_f(ir + hr);
        const float zz = sigmoid_f(iz + hz);
        const float nn = tanh_f(in_ + r * hn);
        const int hp = hoff + fpos(b, j);
        const float hnew = (1.f - zz) * nn + zz * (float)sm[hp];
        sm[hp] = (_Float16)hnew;
        if (DSTK == DST_LDS) sm[ldstOff + fpos(b, dstCol + j)] = (_Float16)hnew;
        if (DSTK == DST_GLB) gdst[fpos(b, dstCol + j)] = (_Float16)hnew;
    }
    __syncthreads();
}

__global__ void __launch_bounds__(32)
vseq2seq_wmma_kernel(KParams P) {
    __shared__ __align__(32) _Float16 smem[SMEM_HALFS];
    __shared__ float s_bias[600];
    __shared__ float s_gi[16 * 32];
    __shared__ float s_gh[16 * 32];

    _Float16* const e3ws = (_Float16*)P.e3ws;
    _Float16* const ysws = (_Float16*)P.ysws;
    const int tile = blockIdx.x;                 // 64 tiles of 16 batch rows
    const int tid = threadIdx.x;
    const int nc = tid & 15;
    v8f z = {};

    // ---------------- zero pool, stage weights / biases ----------------
    for (int i = tid; i < SMEM_HALFS; i += 32) smem[i] = (_Float16)0.f;
    __syncthreads();

    for (int nt = 0; nt < 4; ++nt)
        for (int kt = 0; kt < 2; ++kt)
            stage_w(smem, FRAG(nt * 2 + kt), P.p[1], 64, 64, nt, kt);   // We1
    stage_w(smem, FRAG(8), P.p[3], 10, 64, 0, 0);                        // We2
    stage_w(smem, FRAG(9), P.p[3], 10, 64, 0, 1);
    stage_w(smem, FRAG(10), P.p[5], 10, 10, 0, 0);                       // We3
    {
        const int gbase[4] = {11, 19, 29, 37};
        const int ginp[4]  = {7, 15, 27, 35};
        const int gK[4]    = {10, 20, 10, 20};
        for (int g = 0; g < 4; ++g) {
            const int fb = gbase[g], ib = ginp[g], K = gK[g];
            stage_w(smem, FRAG(fb + 0), P.p[ib + 0], 30, K, 0, 0);   // Wi_f
            stage_w(smem, FRAG(fb + 1), P.p[ib + 0], 30, K, 1, 0);
            stage_w(smem, FRAG(fb + 2), P.p[ib + 1], 30, 10, 0, 0);  // Wh_f
            stage_w(smem, FRAG(fb + 3), P.p[ib + 1], 30, 10, 1, 0);
            stage_w(smem, FRAG(fb + 4), P.p[ib + 4], 30, K, 0, 0);   // Wi_b
            stage_w(smem, FRAG(fb + 5), P.p[ib + 4], 30, K, 1, 0);
            stage_w(smem, FRAG(fb + 6), P.p[ib + 5], 30, 10, 0, 0);  // Wh_b
            stage_w(smem, FRAG(fb + 7), P.p[ib + 5], 30, 10, 1, 0);
        }
    }
    stage_w(smem, FRAG(27), P.p[23], 10, 20, 0, 0);                      // Wd1
    stage_w(smem, FRAG(28), P.p[25], 10, 10, 0, 0);                      // Wd2
    stage_w(smem, FRAG(45), P.p[43], 16, 20, 0, 0);                      // Wo

    copyf(&s_bias[BO_BE1], P.p[2], 64);
    copyf(&s_bias[BO_BE2], P.p[4], 10);
    copyf(&s_bias[BO_BE3], P.p[6], 10);
    copyf(&s_bias[BO_EG0 + 0],  P.p[9], 30);   copyf(&s_bias[BO_EG0 + 30], P.p[10], 30);
    copyf(&s_bias[BO_EG0 + 60], P.p[13], 30);  copyf(&s_bias[BO_EG0 + 90], P.p[14], 30);
    copyf(&s_bias[BO_EG1 + 0],  P.p[17], 30);  copyf(&s_bias[BO_EG1 + 30], P.p[18], 30);
    copyf(&s_bias[BO_EG1 + 60], P.p[21], 30);  copyf(&s_bias[BO_EG1 + 90], P.p[22], 30);
    copyf(&s_bias[BO_BD1], P.p[24], 10);
    copyf(&s_bias[BO_BD2], P.p[26], 10);
    copyf(&s_bias[BO_DG0 + 0],  P.p[29], 30);  copyf(&s_bias[BO_DG0 + 30], P.p[30], 30);
    copyf(&s_bias[BO_DG0 + 60], P.p[33], 30);  copyf(&s_bias[BO_DG0 + 90], P.p[34], 30);
    copyf(&s_bias[BO_DG1 + 0],  P.p[37], 30);  copyf(&s_bias[BO_DG1 + 30], P.p[38], 30);
    copyf(&s_bias[BO_DG1 + 60], P.p[41], 30);  copyf(&s_bias[BO_DG1 + 90], P.p[42], 30);
    copyf(&s_bias[BO_BOUT], P.p[44], 16);
    __syncthreads();

    // ------------- phase 1: embedding fused with enc GRU l0 fwd -------------
    {
        v16h wf[15];
#pragma unroll
        for (int i = 0; i < 15; ++i) wf[i] = load_frag(smem, FRAG(i));   // We1x8,We2x2,We3, l0f cell x4
        float be1v[4];
#pragma unroll
        for (int nt = 0; nt < 4; ++nt) be1v[nt] = s_bias[BO_BE1 + nt * 16 + nc];
        const float be2v = (nc < 10) ? s_bias[BO_BE2 + nc] : 0.f;
        const float be3v = (nc < 10) ? s_bias[BO_BE3 + nc] : 0.f;
        const float bi0 = s_bias[BO_EG0 + nc],      bi1 = bias_hi30(&s_bias[BO_EG0]);
        const float bh0 = s_bias[BO_EG0 + 30 + nc], bh1 = bias_hi30(&s_bias[BO_EG0 + 30]);
        const int khi = (tid >= 16) ? 8 : 0;
        const float* obs = P.p[0];
        for (int t = 0; t < SEQL; ++t) {
            const float* op = obs + ((size_t)t * NBAT + (size_t)tile * 16) * OBSD;
            v16h a0, a1;                                       // obs A-frags straight from global
#pragma unroll
            for (int h = 0; h < 16; ++h) {
                const int k = ((h < 8) ? h : h + 8) + khi;
                a0[h] = (_Float16)op[nc * 64 + k];
                a1[h] = (_Float16)op[nc * 64 + 32 + k];
            }
#pragma unroll
            for (int nt = 0; nt < 4; ++nt) {                   // e1 = lrelu
                v8f d = wmma16(a0, wf[nt * 2 + 0], z);
                d     = wmma16(a1, wf[nt * 2 + 1], d);
                store_frag_act(smem, OF_E1, d, nt, be1v[nt], 64, 1);
            }
            __syncthreads();
            v8f d2 = wmma16(load_frag(smem, OF_E1),       wf[8], z);   // e2
            d2     = wmma16(load_frag(smem, OF_E1 + 512), wf[9], d2);
            store_frag_act(smem, OF_X, d2, 0, be2v, 10, 1);
            __syncthreads();
            v8f d3 = wmma16(load_frag(smem, OF_X), wf[10], z);         // e3 = tanh
            store_frag_act(smem, OF_E, d3, 0, be3v, 10, 2);
            __syncthreads();
            v16h axv = load_frag(smem, OF_E);
            *(v16h*)(e3ws + ((size_t)tile * SEQL + t) * 512 + (size_t)tid * 16) = axv;
            gru_cell<DST_GLB>(smem, axv, OF_H, wf[11], wf[12], wf[13], wf[14],
                              bi0, bi1, bh0, bh1, s_gi, s_gh, 0, 0,
                              ysws + ((size_t)tile * SEQL + t) * 512);
        }
    }

    // ------------- phase 2: enc GRU l0 bwd -------------
    {
        v16h w0 = load_frag(smem, FRAG(15)), w1 = load_frag(smem, FRAG(16));
        v16h w2 = load_frag(smem, FRAG(17)), w3 = load_frag(smem, FRAG(18));
        const float bi0 = s_bias[BO_EG0 + 60 + nc], bi1 = bias_hi30(&s_bias[BO_EG0 + 60]);
        const float bh0 = s_bias[BO_EG0 + 90 + nc], bh1 = bias_hi30(&s_bias[BO_EG0 + 90]);
        for (int t = SEQL - 1; t >= 0; --t) {
            v16h axv = *(const v16h*)(e3ws + ((size_t)tile * SEQL + t) * 512 + (size_t)tid * 16);
            gru_cell<DST_GLB>(smem, axv, OF_H + 512, w0, w1, w2, w3,
                              bi0, bi1, bh0, bh1, s_gi, s_gh, 0, 10,
                              ysws + ((size_t)tile * SEQL + t) * 512);
        }
    }

    // ------------- phase 3/4: enc GRU l1 fwd + bwd (final h only) -------------
    {
        v16h w0 = load_frag(smem, FRAG(19)), w1 = load_frag(smem, FRAG(20));
        v16h w2 = load_frag(smem, FRAG(21)), w3 = load_frag(smem, FRAG(22));
        const float bi0 = s_bias[BO_EG1 + nc],      bi1 = bias_hi30(&s_bias[BO_EG1]);
        const float bh0 = s_bias[BO_EG1 + 30 + nc], bh1 = bias_hi30(&s_bias[BO_EG1 + 30]);
        for (int t = 0; t < SEQL; ++t) {
            v16h axv = *(const v16h*)(ysws + ((size_t)tile * SEQL + t) * 512 + (size_t)tid * 16);
            gru_cell<DST_NONE>(smem, axv, OF_H + 1024, w0, w1, w2, w3,
                               bi0, bi1, bh0, bh1, s_gi, s_gh, 0, 0, nullptr);
        }
    }
    {
        v16h w0 = load_frag(smem, FRAG(23)), w1 = load_frag(smem, FRAG(24));
        v16h w2 = load_frag(smem, FRAG(25)), w3 = load_frag(smem, FRAG(26));
        const float bi0 = s_bias[BO_EG1 + 60 + nc], bi1 = bias_hi30(&s_bias[BO_EG1 + 60]);
        const float bh0 = s_bias[BO_EG1 + 90 + nc], bh1 = bias_hi30(&s_bias[BO_EG1 + 90]);
        for (int t = SEQL - 1; t >= 0; --t) {
            v16h axv = *(const v16h*)(ysws + ((size_t)tile * SEQL + t) * 512 + (size_t)tid * 16);
            gru_cell<DST_NONE>(smem, axv, OF_H + 1536, w0, w1, w2, w3,
                               bi0, bi1, bh0, bh1, s_gi, s_gh, 0, 0, nullptr);
        }
    }

    // ------------- phase 5: decoder -------------
    for (int i = tid; i < 16 * 32; i += 32) smem[OF_OUT + i] = (_Float16)0.f;  // out0 = 0
    for (int i = tid; i < 256; i += 32)                                         // z_hat[0] = 0
        P.out[((size_t)tile * 16 + (i >> 4)) * 16 + (i & 15)] = 0.f;
    __syncthreads();
    {
        v16h wd1 = load_frag(smem, FRAG(27));
        v16h wd2 = load_frag(smem, FRAG(28));
        v16h wo  = load_frag(smem, FRAG(45));
        v16h c[16];
#pragma unroll
        for (int i = 0; i < 16; ++i) c[i] = load_frag(smem, FRAG(29 + i));  // dec l0 x8, l1 x8
        const float bd1v = (nc < 10) ? s_bias[BO_BD1 + nc] : 0.f;
        const float bd2v = (nc < 10) ? s_bias[BO_BD2 + nc] : 0.f;
        const float b0i0 = s_bias[BO_DG0 + nc],      b0i1 = bias_hi30(&s_bias[BO_DG0]);
        const float b0h0 = s_bias[BO_DG0 + 30 + nc], b0h1 = bias_hi30(&s_bias[BO_DG0 + 30]);
        const float b1i0 = s_bias[BO_DG0 + 60 + nc], b1i1 = bias_hi30(&s_bias[BO_DG0 + 60]);
        const float b1h0 = s_bias[BO_DG0 + 90 + nc], b1h1 = bias_hi30(&s_bias[BO_DG0 + 90]);
        const float b2i0 = s_bias[BO_DG1 + nc],      b2i1 = bias_hi30(&s_bias[BO_DG1]);
        const float b2h0 = s_bias[BO_DG1 + 30 + nc], b2h1 = bias_hi30(&s_bias[BO_DG1 + 30]);
        const float b3i0 = s_bias[BO_DG1 + 60 + nc], b3i1 = bias_hi30(&s_bias[BO_DG1 + 60]);
        const float b3h0 = s_bias[BO_DG1 + 90 + nc], b3h1 = bias_hi30(&s_bias[BO_DG1 + 90]);
        const float bvout = s_bias[BO_BOUT + nc];
        const int mb = (tid >= 16) ? 8 : 0;

        for (int t = 1; t < SEQL; ++t) {
            v8f e1d = wmma16(load_frag(smem, OF_OUT), wd1, z);          // dec emb
            store_frag_act(smem, OF_E, e1d, 0, bd1v, 10, 1);
            __syncthreads();
            v8f e2d = wmma16(load_frag(smem, OF_E), wd2, z);
            store_frag_act(smem, OF_X, e2d, 0, bd2v, 10, 1);
            __syncthreads();
            gru_cell<DST_LDS>(smem, load_frag(smem, OF_X),  OF_H,        c[0],  c[1],  c[2],  c[3],
                              b0i0, b0i1, b0h0, b0h1, s_gi, s_gh, OF_X2, 0, nullptr);
            gru_cell<DST_LDS>(smem, load_frag(smem, OF_X),  OF_H + 512,  c[4],  c[5],  c[6],  c[7],
                              b1i0, b1i1, b1h0, b1h1, s_gi, s_gh, OF_X2, 10, nullptr);
            gru_cell<DST_LDS>(smem, load_frag(smem, OF_X2), OF_H + 1024, c[8],  c[9],  c[10], c[11],
                              b2i0, b2i1, b2h0, b2h1, s_gi, s_gh, OF_OUT, 0, nullptr);
            gru_cell<DST_LDS>(smem, load_frag(smem, OF_X2), OF_H + 1536, c[12], c[13], c[14], c[15],
                              b3i0, b3i1, b3h0, b3h1, s_gi, s_gh, OF_OUT, 10, nullptr);
            v8f pd = wmma16(load_frag(smem, OF_OUT), wo, z);            // fc_out
#pragma unroll
            for (int r = 0; r < 8; ++r)
                P.out[((size_t)t * NBAT + (size_t)tile * 16 + mb + r) * 16 + nc] = pd[r] + bvout;
        }
    }
}

extern "C" void kernel_launch(void* const* d_in, const int* in_sizes, int n_in,
                              void* d_out, int out_size, void* d_ws, size_t ws_size,
                              hipStream_t stream) {
    (void)in_sizes; (void)out_size; (void)ws_size;
    KParams P;
    for (int i = 0; i < 45; ++i) P.p[i] = (i < n_in) ? (const float*)d_in[i] : nullptr;
    P.out = (float*)d_out;
    P.e3ws = d_ws;                                                        // 33,554,432 B
    P.ysws = (void*)((char*)d_ws + (size_t)64 * SEQL * 512 * sizeof(short));
    vseq2seq_wmma_kernel<<<dim3(64), dim3(32), 0, stream>>>(P);
}
